// KerasLlamaAttention_2791728742713
// MI455X (gfx1250) — matrix operand order repaired and checked by reading, not compiled, not run on hardware
//
#include <hip/hip_runtime.h>
#include <math.h>

#define HIDDEN 2048
#define NH 32
#define NKV 8
#define HD 64
#define SEQN 2048
#define BS 2

typedef __attribute__((ext_vector_type(16))) __bf16 v16bf;
typedef __attribute__((ext_vector_type(4)))  __bf16 v4bf;
typedef __attribute__((ext_vector_type(8)))  float  v8f;
typedef __attribute__((ext_vector_type(4)))  unsigned int u32x4;
typedef __attribute__((ext_vector_type(8)))  int i32x8;
typedef __attribute__((ext_vector_type(4)))  int i32x4;

__device__ __forceinline__ __bf16 f2bf(float x) { return (__bf16)x; }

// A fragment: 16(M) x 32(K) row-major tile in LDS (ISA 7.12.2 16-bit A layout)
__device__ __forceinline__ v16bf load_a_lds(const __bf16* base, int stride, int lane) {
  int r = lane & 15, h = lane >> 4;
  v16bf f;
#pragma unroll
  for (int e = 0; e < 16; ++e) {
    int k = (e < 8) ? (8 * h + e) : (e + 8 + 8 * h);
    f[e] = base[r * stride + k];
  }
  return f;
}

// B fragment from transposed (N=16 rows) x (K=32) row-major tile: table K order
// (8-element K chunks at 8h and 16+8h per lane half), matching the A fragment.
__device__ __forceinline__ v16bf load_b_lds(const __bf16* base, int stride, int lane) {
  int n = lane & 15, h = lane >> 4;
  v16bf f;
#pragma unroll
  for (int e = 0; e < 16; ++e) f[e] = base[n * stride + ((e < 8) ? (8 * h + e) : (e + 8 + 8 * h))];
  return f;
}

// ---------------------------------------------------------------------------
// TDM: 2D tile load global -> LDS with LDS row padding done by the DMA.
// D# per CDNA5 ISA ch.8. group0 = {count, lds_addr, global_addr, type=2};
// group1 = {dsize/pad, tensor dims, tile dims, dim0 stride}; groups 2/3 and
// the extra vector are zero (2D tile). This toolchain's builtin takes 6 args.
// pad_interval code v => 2^(v+1) DWORDs between pads; pad_amount code v => v+1 DWORDs.
// ---------------------------------------------------------------------------
__device__ __forceinline__ void tdm_load_2d(
    unsigned int lds_byte, const void* gptr, unsigned int tile_w_elems,
    unsigned int tile_h, unsigned int stride_elems,
    unsigned int pad_int_code, unsigned int pad_amt_code) {
  unsigned long long ga = (unsigned long long)(uintptr_t)gptr;
  u32x4 g0;
  g0[0] = 1u;                                   // count=1, user mode
  g0[1] = lds_byte;                             // lds_addr (bytes)
  g0[2] = (unsigned int)(ga & 0xffffffffu);     // global_addr[31:0]
  g0[3] = (unsigned int)((ga >> 32) & 0x1ffffffu) | (2u << 30);  // addr[56:32]|type=2
  i32x8 g1;
  g1[0] = (int)((1u << 16)                      // data_size = 2 bytes
              | (1u << 20)                      // pad_enable
              | (pad_int_code << 22)
              | (pad_amt_code << 25));
  g1[1] = (int)(tile_w_elems << 16);            // tensor_dim0 (lo16)
  g1[2] = (int)(tile_h << 16);                  // tensor_dim1 (lo16); dim0 hi = 0
  g1[3] = (int)(tile_w_elems << 16);            // tile_dim0; dim1 hi = 0
  g1[4] = (int)tile_h;                          // tile_dim1; tile_dim2 = 0
  g1[5] = (int)stride_elems;                    // tensor_dim0_stride[31:0]
  g1[6] = 0;
  g1[7] = 0;
  i32x4 gz4 = {0, 0, 0, 0};
  i32x8 gz8 = {0, 0, 0, 0, 0, 0, 0, 0};
  __builtin_amdgcn_tensor_load_to_lds(g0, g1, gz4, gz4, gz8, 0);
}

// ---------------------------------------------------------------------------
// One-time prep: fp32 -> bf16 convert, and fp32 [K][N] -> bf16 [N][K] transpose
// ---------------------------------------------------------------------------
__global__ void cvt_bf16(const float* __restrict__ X, __bf16* __restrict__ Y, int n4) {
  int i = blockIdx.x * blockDim.x + threadIdx.x;
  if (i >= n4) return;
  float4 v = ((const float4*)X)[i];
  v4bf o;
  o[0] = f2bf(v.x); o[1] = f2bf(v.y); o[2] = f2bf(v.z); o[3] = f2bf(v.w);
  ((v4bf*)Y)[i] = o;
}

__global__ __launch_bounds__(256) void transpose_cvt(
    const float* __restrict__ W, __bf16* __restrict__ Wt, int K, int N) {
  __shared__ __bf16 tile[64][72];
  int k0 = blockIdx.y * 64, n0 = blockIdx.x * 64;
  int tc = threadIdx.x & 63, tr4 = threadIdx.x >> 6;
#pragma unroll
  for (int i = 0; i < 16; ++i) {
    int r = tr4 * 16 + i;
    tile[r][tc] = f2bf(W[(size_t)(k0 + r) * N + n0 + tc]);
  }
  __syncthreads();
#pragma unroll
  for (int i = 0; i < 16; ++i) {
    int r = tr4 * 16 + i;  // local n
    Wt[(size_t)(n0 + r) * K + k0 + tc] = tile[tc][r];
  }
}

// ---------------------------------------------------------------------------
// GEMM: Out = A[M,K](bf16,K-major) * Wt[N,K](bf16,K-major), WMMA bf16->f32.
// Tiles staged by TDM, double-buffered with split tensorcnt waits.
// EPI 0: fp32 out; EPI 1: bf16 out + RoPE; EPI 2: bf16 V transposed out.
// ---------------------------------------------------------------------------
#define APAD 40
template <int EPI>
__global__ __launch_bounds__(128) void gemm_k(
    const __bf16* __restrict__ A, const __bf16* __restrict__ B,
    void* __restrict__ OutP, int M, int N, int K) {
  __shared__ __bf16 As[2][64 * APAD];
  __shared__ __bf16 Bs[2][64 * APAD];
  int t = threadIdx.x, lane = t & 31, w = t >> 5;
  int M0 = blockIdx.y * 64, N0 = blockIdx.x * 64;
  int r0 = t >> 2, c0 = (t & 3) * 8;   // for prefetch addressing only

  v8f zero = {};
  v8f acc[4];
#pragma unroll
  for (int i = 0; i < 4; ++i) acc[i] = zero;

  const __bf16* ap = A + (size_t)(M0 + r0) * K + c0;
  const __bf16* bp = B + (size_t)(N0 + r0) * K + c0;
  const __bf16* Abase = A + (size_t)M0 * K;
  const __bf16* Bbase = B + (size_t)N0 * K;

  int nIter = K >> 5;
  if (t < 32) {  // wave 0 drives the DMA (per-wave TENSORcnt)
    tdm_load_2d((unsigned int)(uintptr_t)&As[0][0], Abase, 32, 64, K, 3, 3);
    tdm_load_2d((unsigned int)(uintptr_t)&Bs[0][0], Bbase, 32, 64, K, 3, 3);
  }
  for (int i = 0; i < nIter; ++i) {
    int cur = i & 1;
    if (i + 1 < nIter) {  // warm GL2 for the next DMA tile
      __builtin_prefetch(ap + (i + 1) * 32, 0, 0);
      __builtin_prefetch(bp + (i + 1) * 32, 0, 0);
    }
    if (t < 32) {
      if (i + 1 < nIter) {
        int k1 = (i + 1) * 32;
        tdm_load_2d((unsigned int)(uintptr_t)&As[1 - cur][0], Abase + k1, 32, 64, K, 3, 3);
        tdm_load_2d((unsigned int)(uintptr_t)&Bs[1 - cur][0], Bbase + k1, 32, 64, K, 3, 3);
        __builtin_amdgcn_s_wait_tensorcnt(2);  // current tile done; next 2 in flight
      } else {
        __builtin_amdgcn_s_wait_tensorcnt(0);
      }
    }
    __syncthreads();
    v16bf a = load_a_lds(&As[cur][w * 16 * APAD], APAD, lane);
#pragma unroll
    for (int nt = 0; nt < 4; ++nt) {
      v16bf b = load_b_lds(&Bs[cur][nt * 16 * APAD], APAD, lane);
      acc[nt] = __builtin_amdgcn_wmma_f32_16x16x32_bf16(
          false, a, false, b, (short)0, acc[nt], false, false);
    }
    __syncthreads();
  }

  int half = lane >> 4, col = lane & 15;
  if constexpr (EPI == 0) {
    float* Out = (float*)OutP;
#pragma unroll
    for (int nt = 0; nt < 4; ++nt)
#pragma unroll
      for (int r = 0; r < 8; ++r) {
        int m = M0 + w * 16 + r + 8 * half;
        Out[(size_t)m * N + N0 + nt * 16 + col] = acc[nt][r];
      }
  } else if constexpr (EPI == 1) {
    __bf16* Out = (__bf16*)OutP;
#pragma unroll
    for (int nt = 0; nt < 4; ++nt) {
      int d = nt * 16 + col;      // head-local dim (N0 is a multiple of 64)
      int j = d & 31;
      float ifr = __powf(10000.0f, -(float)j * (1.0f / 32.0f));
#pragma unroll
      for (int r = 0; r < 8; ++r) {
        int m = M0 + w * 16 + r + 8 * half;
        float ang = (float)(m & (SEQN - 1)) * ifr;
        float si, c;
        __sincosf(ang, &si, &c);
        float x = acc[nt][r], p = acc[nt ^ 2][r];  // rotate_half partner
        float v = (d < 32) ? (x * c - p * si) : (x * c + p * si);
        Out[(size_t)m * N + N0 + d] = f2bf(v);
      }
    }
  } else {  // EPI == 2: V transposed to [b][kvh][d][s]
    __bf16* Out = (__bf16*)OutP;
    int b = M0 >> 11;  // M0 / SEQN
    int s0 = (M0 & (SEQN - 1)) + w * 16 + 8 * half;
#pragma unroll
    for (int nt = 0; nt < 4; ++nt) {
      int n = N0 + nt * 16 + col;
      int kvh = n >> 6, d = n & 63;
      v4bf o0, o1;
#pragma unroll
      for (int rr = 0; rr < 4; ++rr) {
        o0[rr] = f2bf(acc[nt][rr]);
        o1[rr] = f2bf(acc[nt][rr + 4]);
      }
      __bf16* dst = Out + ((size_t)(b * NKV + kvh) * HD + d) * SEQN + s0;
      *(v4bf*)dst = o0;
      *(v4bf*)(dst + 4) = o1;
    }
  }
}

// ---------------------------------------------------------------------------
// Flash attention: grid (S/64, H, B), 128 threads (4 waves), online softmax.
// Q bf16 [b][s][h*64+d]; K bf16 [b][s][kvh*64+d]; V bf16 [b][kvh][d][s].
// K/V tiles staged by TDM, double-buffered.
// ---------------------------------------------------------------------------
#define KPAD 72
__global__ __launch_bounds__(128) void flash_attn(
    const __bf16* __restrict__ Q, const __bf16* __restrict__ Kb,
    const __bf16* __restrict__ Vb, __bf16* __restrict__ O) {
  __shared__ __bf16 Kt[2][64 * KPAD];  // [kv][d]  (B^T for Q*K^T)
  __shared__ __bf16 Vt[2][64 * KPAD];  // [d][kv]  (B^T for P*V)
  __shared__ __bf16 Pt[64 * KPAD];     // [q][kv]  (A for P*V)
  int t = threadIdx.x, lane = t & 31, w = t >> 5;
  int half = lane >> 4, col = lane & 15;
  int qt = blockIdx.x, h = blockIdx.y, b = blockIdx.z;
  int kvh = h >> 2;
  int q0 = qt * 64;

  const __bf16* Kbase0 = Kb + (size_t)(b * SEQN) * (NKV * HD) + kvh * HD;
  const __bf16* Vbase0 = Vb + (size_t)(b * NKV + kvh) * HD * SEQN;

  if (t < 32) {  // preload first kv tile
    tdm_load_2d((unsigned int)(uintptr_t)&Kt[0][0], Kbase0, 64, 64, NKV * HD, 4, 3);
    tdm_load_2d((unsigned int)(uintptr_t)&Vt[0][0], Vbase0, 64, 64, SEQN, 4, 3);
  }

  // Q fragments, resident in registers for the whole kernel
  v16bf aq0, aq1;
  {
    const __bf16* qp =
        Q + (size_t)(b * SEQN + q0 + w * 16 + col) * HIDDEN + h * HD;
#pragma unroll
    for (int e = 0; e < 16; ++e) {
      int k = (e < 8) ? (8 * half + e) : (e + 8 + 8 * half);
      aq0[e] = qp[k];
      aq1[e] = qp[32 + k];
    }
  }

  v8f zero = {};
  v8f oacc[4];
#pragma unroll
  for (int i = 0; i < 4; ++i) oacc[i] = zero;
  float mst[8], lst[8];
#pragma unroll
  for (int r = 0; r < 8; ++r) { mst[r] = -1e30f; lst[r] = 0.0f; }
  int qrow_base = q0 + w * 16 + 8 * half;

  for (int jt = 0; jt <= qt; ++jt) {
    int cur = jt & 1;
    int j0 = jt * 64;
    if (t < 32) {
      if (jt < qt) {
        int j1 = (jt + 1) * 64;
        tdm_load_2d((unsigned int)(uintptr_t)&Kt[1 - cur][0],
                    Kbase0 + (size_t)j1 * (NKV * HD), 64, 64, NKV * HD, 4, 3);
        tdm_load_2d((unsigned int)(uintptr_t)&Vt[1 - cur][0],
                    Vbase0 + j1, 64, 64, SEQN, 4, 3);
        __builtin_amdgcn_s_wait_tensorcnt(2);
      } else {
        __builtin_amdgcn_s_wait_tensorcnt(0);
      }
    }
    __syncthreads();

    // S = Q * K^T
    v8f sacc[4];
#pragma unroll
    for (int nt = 0; nt < 4; ++nt) {
      v8f z = zero;
      v16bf b0 = load_b_lds(&Kt[cur][nt * 16 * KPAD], KPAD, lane);
      v16bf b1 = load_b_lds(&Kt[cur][nt * 16 * KPAD + 32], KPAD, lane);
      z = __builtin_amdgcn_wmma_f32_16x16x32_bf16(false, aq0, false, b0,
                                                  (short)0, z, false, false);
      z = __builtin_amdgcn_wmma_f32_16x16x32_bf16(false, aq1, false, b1,
                                                  (short)0, z, false, false);
      sacc[nt] = z;
    }

    // Online softmax (xor masks 1,2,4,8 stay inside each 16-lane half)
#pragma unroll
    for (int r = 0; r < 8; ++r) {
      int qrow = qrow_base + r;
      float sv[4];
      float mx = -1e30f;
#pragma unroll
      for (int nt = 0; nt < 4; ++nt) {
        float s = sacc[nt][r] * 0.125f;
        int kv = j0 + nt * 16 + col;
        if (kv > qrow) s = -1e30f;
        sv[nt] = s;
        mx = fmaxf(mx, s);
      }
      mx = fmaxf(mx, __shfl_xor(mx, 1, 32));
      mx = fmaxf(mx, __shfl_xor(mx, 2, 32));
      mx = fmaxf(mx, __shfl_xor(mx, 4, 32));
      mx = fmaxf(mx, __shfl_xor(mx, 8, 32));
      float mnew = fmaxf(mst[r], mx);
      float corr = __expf(mst[r] - mnew);
      float psum = 0.0f;
#pragma unroll
      for (int nt = 0; nt < 4; ++nt) {
        float p = __expf(sv[nt] - mnew);
        psum += p;
        Pt[(w * 16 + r + 8 * half) * KPAD + nt * 16 + col] = f2bf(p);
      }
      psum += __shfl_xor(psum, 1, 32);
      psum += __shfl_xor(psum, 2, 32);
      psum += __shfl_xor(psum, 4, 32);
      psum += __shfl_xor(psum, 8, 32);
      lst[r] = lst[r] * corr + psum;
      mst[r] = mnew;
#pragma unroll
      for (int dt = 0; dt < 4; ++dt) oacc[dt][r] *= corr;
    }

    // O += P * V (Pt written+read by the same wave; DS is in-order per wave)
    v16bf ap0 = load_a_lds(&Pt[w * 16 * KPAD], KPAD, lane);
    v16bf ap1 = load_a_lds(&Pt[w * 16 * KPAD + 32], KPAD, lane);
#pragma unroll
    for (int dt = 0; dt < 4; ++dt) {
      v16bf b0 = load_b_lds(&Vt[cur][dt * 16 * KPAD], KPAD, lane);
      v16bf b1 = load_b_lds(&Vt[cur][dt * 16 * KPAD + 32], KPAD, lane);
      oacc[dt] = __builtin_amdgcn_wmma_f32_16x16x32_bf16(
          false, ap0, false, b0, (short)0, oacc[dt], false, false);
      oacc[dt] = __builtin_amdgcn_wmma_f32_16x16x32_bf16(
          false, ap1, false, b1, (short)0, oacc[dt], false, false);
    }
    __syncthreads();
  }

  // Normalize, store bf16 attn output in [b][s][h*64+d] for the wo GEMM
#pragma unroll
  for (int dt = 0; dt < 4; ++dt)
#pragma unroll
    for (int r = 0; r < 8; ++r) {
      int srow = q0 + w * 16 + r + 8 * half;
      float inv = 1.0f / lst[r];
      O[(size_t)(b * SEQN + srow) * HIDDEN + h * HD + dt * 16 + col] =
          f2bf(oacc[dt][r] * inv);
    }
}

// ---------------------------------------------------------------------------
extern "C" void kernel_launch(void* const* d_in, const int* in_sizes, int n_in,
                              void* d_out, int out_size, void* d_ws,
                              size_t ws_size, hipStream_t stream) {
  const float* hs = (const float*)d_in[0];
  const float* wq = (const float*)d_in[1];
  const float* wk = (const float*)d_in[2];
  const float* wv = (const float*)d_in[3];
  const float* wo = (const float*)d_in[4];
  // d_in[5] (causal mask) / d_in[6] (position_ids=arange) realized analytically
  float* out = (float*)d_out;

  const int M = BS * SEQN;  // 4096
  const int KVD = NKV * HD; // 512
  __bf16* Xb  = (__bf16*)d_ws;
  __bf16* Wtq = Xb  + (size_t)M * HIDDEN;
  __bf16* Wtk = Wtq + (size_t)HIDDEN * HIDDEN;
  __bf16* Wtv = Wtk + (size_t)KVD * HIDDEN;
  __bf16* Wto = Wtv + (size_t)KVD * HIDDEN;
  __bf16* Qb  = Wto + (size_t)HIDDEN * HIDDEN;
  __bf16* Kb  = Qb  + (size_t)M * HIDDEN;
  __bf16* Vtb = Kb  + (size_t)M * KVD;
  __bf16* Ab  = Vtb + (size_t)M * KVD;

  dim3 blk(128);
  // Prep: activations to bf16, weights to bf16 [N][K]
  int n4 = M * HIDDEN / 4;
  cvt_bf16<<<(n4 + 255) / 256, 256, 0, stream>>>(hs, Xb, n4);
  transpose_cvt<<<dim3(HIDDEN / 64, HIDDEN / 64), 256, 0, stream>>>(wq, Wtq, HIDDEN, HIDDEN);
  transpose_cvt<<<dim3(KVD / 64, HIDDEN / 64), 256, 0, stream>>>(wk, Wtk, HIDDEN, KVD);
  transpose_cvt<<<dim3(KVD / 64, HIDDEN / 64), 256, 0, stream>>>(wv, Wtv, HIDDEN, KVD);
  transpose_cvt<<<dim3(HIDDEN / 64, HIDDEN / 64), 256, 0, stream>>>(wo, Wto, HIDDEN, HIDDEN);
  // Projections (RoPE fused into Q/K epilogues; V written pre-transposed)
  gemm_k<1><<<dim3(HIDDEN / 64, M / 64), blk, 0, stream>>>(Xb, Wtq, Qb, M, HIDDEN, HIDDEN);
  gemm_k<1><<<dim3(KVD / 64, M / 64), blk, 0, stream>>>(Xb, Wtk, Kb, M, KVD, HIDDEN);
  gemm_k<2><<<dim3(KVD / 64, M / 64), blk, 0, stream>>>(Xb, Wtv, Vtb, M, KVD, HIDDEN);
  // Flash attention
  flash_attn<<<dim3(SEQN / 64, NH, BS), blk, 0, stream>>>(Qb, Kb, Vtb, Ab);
  // Output projection -> fp32
  gemm_k<0><<<dim3(HIDDEN / 64, M / 64), blk, 0, stream>>>(Ab, Wto, out, M, HIDDEN, HIDDEN);
}
